// LiteMLA3D_9380208575218
// MI455X (gfx1250) — matrix operand order repaired and checked by reading, not compiled, not run on hardware
//
#include <hip/hip_runtime.h>
#include <hip/hip_bf16.h>

#define DEVINL __device__ __forceinline__

typedef __attribute__((ext_vector_type(16))) __bf16        v16bf;
typedef __attribute__((ext_vector_type(8)))  float         v8f;
typedef __attribute__((ext_vector_type(16))) unsigned short us16;

namespace {
constexpr int Bn = 4;        // batch
constexpr int C  = 256;      // channels
constexpr int Nn = 16384;    // D*H*W
constexpr int Vp = 272;      // 257 (C + ones column) padded to 17*16
constexpr int M3 = 768;      // 3*C qkv rows
}

struct U4x2 { uint4 a, b; };

// native f32 -> bf16 (gfx1250 has hardware bf16 converts; single VALU op)
DEVINL unsigned short f2bf(float f) {
  __bf16 h = (__bf16)f;
  return __builtin_bit_cast(unsigned short, h);
}

// A-matrix 16x32 bf16 fragment (this code's contiguous k order): lane holds row M=lane&15,
// slot s -> k = s + 16*hi. Two contiguous 16B chunks per lane.
DEVINL v16bf afrag(const unsigned short* row, int kbase, int hi) {
  U4x2 t;
  t.a = *reinterpret_cast<const uint4*>(row + kbase + 16 * hi);
  t.b = *reinterpret_cast<const uint4*>(row + kbase + 16 * hi + 8);
  return __builtin_bit_cast(v16bf, t);
}

// B-matrix 32x16 bf16 fragment: lane holds column N=lane&15,
// slot s -> k = s + 16*hi. One contiguous 32B run per lane (k contiguous in storage).
DEVINL v16bf bfrag(const unsigned short* col, int kbase, int hi) {
  const unsigned short* p = col + kbase + 16 * hi;
  U4x2 t;
  t.a = *reinterpret_cast<const uint4*>(p);
  t.b = *reinterpret_cast<const uint4*>(p + 8);
  return __builtin_bit_cast(v16bf, t);
}

DEVINL v8f wmma_bf16(v16bf a, v16bf b, v8f c) {
  return __builtin_amdgcn_wmma_f32_16x16x32_bf16(false, a, false, b, (short)0, c,
                                                 false, false);
}

// ---------------------------------------------------------------------------
// K0: fold BN into proj weight/bias, convert weights to bf16, init vpT ones
//     rows + zero kv accumulator.
// ---------------------------------------------------------------------------
__global__ void prep_kernel(const float* __restrict__ Wqkv, const float* __restrict__ Wproj,
                            const float* __restrict__ gamma, const float* __restrict__ beta,
                            const float* __restrict__ mean, const float* __restrict__ var,
                            unsigned short* __restrict__ Wq_bf,
                            unsigned short* __restrict__ Wp_bf,
                            float* __restrict__ bias,
                            unsigned short* __restrict__ vpT,
                            float* __restrict__ kvT) {
  const long R0 = (long)M3 * C;        // W_qkv elements
  const long R1 = (long)C * C;         // W_proj elements
  const long R2 = C;                   // bias
  const long R3 = (long)Bn * 16 * Nn;  // vpT rows 256..271
  const long R4 = (long)Bn * Vp * C;   // kvT zero
  const long TOTAL = R0 + R1 + R2 + R3 + R4;
  for (long i = (long)blockIdx.x * blockDim.x + threadIdx.x; i < TOTAL;
       i += (long)gridDim.x * blockDim.x) {
    long t = i;
    if (t < R0) { Wq_bf[t] = f2bf(Wqkv[t]); continue; }
    t -= R0;
    if (t < R1) {
      int o = (int)(t >> 8);
      float inv = gamma[o] * rsqrtf(var[o] + 1e-5f);
      Wp_bf[t] = f2bf(Wproj[t] * inv);
      continue;
    }
    t -= R1;
    if (t < R2) {
      float inv = gamma[t] * rsqrtf(var[t] + 1e-5f);
      bias[t] = beta[t] - mean[t] * inv;
      continue;
    }
    t -= R2;
    if (t < R3) {
      long b = t / (16L * Nn);
      long rem = t - b * 16L * Nn;
      long row = 256 + rem / Nn;
      long n = rem % Nn;
      vpT[(b * Vp + row) * Nn + n] = (row == 256) ? (unsigned short)0x3F80u
                                                  : (unsigned short)0;
      continue;
    }
    t -= R3;
    kvT[t] = 0.0f;
  }
}

// ---------------------------------------------------------------------------
// K1: qkv = W_qkv @ x  (per batch 768x256x16384), ReLU on q/k,
//     store q:(N,C), k:(C,N), v:vpT rows 0..255 of (272,N), all bf16.
//     Block: 128(M) x 128(N); 8 waves of 2x4 16x16 tiles; K-step 32.
// ---------------------------------------------------------------------------
__global__ __launch_bounds__(256) void qkv_kernel(
    const float* __restrict__ x, const unsigned short* __restrict__ Wq,
    unsigned short* __restrict__ q_s, unsigned short* __restrict__ k_s,
    unsigned short* __restrict__ vpT) {
  const int b = blockIdx.z;
  const int w = threadIdx.x >> 5;
  const int l = threadIdx.x & 31;
  const int ln = l & 15, hi = l >> 4;
  const int wm = blockIdx.y * 128 + ((w >> 1) << 5);  // M base for this wave
  const int wn = blockIdx.x * 128 + ((w & 1) << 6);   // N base for this wave

  const float* xb = x + (size_t)b * C * Nn;
  const unsigned short* arow0 = Wq + (size_t)(wm + ln) * C;
  const unsigned short* arow1 = Wq + (size_t)(wm + 16 + ln) * C;

  v8f acc[2][4] = {};
  for (int kb = 0; kb < C; kb += 32) {
    v16bf a0 = afrag(arow0, kb, hi);
    v16bf a1 = afrag(arow1, kb, hi);
#pragma unroll
    for (int j = 0; j < 4; ++j) {
      // B = x[b] (c,n): per lane column n, k=c strided by Nn (coalesced across lanes)
      const float* xp = xb + (size_t)(kb + 16 * hi) * Nn + (wn + 16 * j + ln);
      us16 uv;
#pragma unroll
      for (int s = 0; s < 16; ++s) uv[s] = f2bf(xp[(size_t)s * Nn]);
      v16bf bf = __builtin_bit_cast(v16bf, uv);
      acc[0][j] = wmma_bf16(a0, bf, acc[0][j]);
      acc[1][j] = wmma_bf16(a1, bf, acc[1][j]);
    }
  }

  // torch reshape(B,C,3,N): row o -> channel c=o/3, kind=o%3 (0=q,1=k,2=v)
#pragma unroll
  for (int i = 0; i < 2; ++i) {
#pragma unroll
    for (int j = 0; j < 4; ++j) {
      const int n = wn + 16 * j + ln;
#pragma unroll
      for (int r = 0; r < 8; ++r) {
        const int o = wm + 16 * i + r + 8 * hi;
        const int c = o / 3;
        const int kind = o - 3 * c;
        const float val = acc[i][j][r];
        if (kind == 0) {
          q_s[((size_t)b * Nn + n) * C + c] = f2bf(fmaxf(val, 0.0f));
        } else if (kind == 1) {
          k_s[((size_t)b * C + c) * Nn + n] = f2bf(fmaxf(val, 0.0f));
        } else {
          vpT[((size_t)b * Vp + c) * Nn + n] = f2bf(val);
        }
      }
    }
  }
}

// ---------------------------------------------------------------------------
// K2: kv[c,v] = sum_n k[c,n]*vp[n,v]  -> stored transposed kvT:(272,256) f32.
//     Split-K over n (16 chunks of 1024), f32 atomic accumulation.
//     Block: 4 waves, each wave one 16-row M tile x 17 v tiles.
// ---------------------------------------------------------------------------
__global__ __launch_bounds__(128) void kv_kernel(
    const unsigned short* __restrict__ k_s, const unsigned short* __restrict__ vpT,
    float* __restrict__ kvT) {
  const int b = blockIdx.z;
  const int l = threadIdx.x & 31;
  const int ln = l & 15, hi = l >> 4;
  const int mbase = blockIdx.y * 64 + ((threadIdx.x >> 5) << 4);
  const int k0 = blockIdx.x * 1024;

  const unsigned short* arow = k_s + ((size_t)b * C + mbase + ln) * Nn;
  const unsigned short* vb = vpT + (size_t)b * Vp * Nn;

  v8f acc[17] = {};
  for (int kk = k0; kk < k0 + 1024; kk += 32) {
    v16bf a = afrag(arow, kk, hi);  // reused across 17 WMMAs
#pragma unroll
    for (int j = 0; j < 17; ++j) {
      v16bf bf = bfrag(vb + (size_t)(16 * j + ln) * Nn, kk, hi);
      acc[j] = wmma_bf16(a, bf, acc[j]);
    }
  }

  float* kvb = kvT + (size_t)b * Vp * C;
#pragma unroll
  for (int j = 0; j < 17; ++j) {
#pragma unroll
    for (int r = 0; r < 8; ++r) {
      atomicAdd(&kvb[(size_t)(16 * j + ln) * C + (mbase + r + 8 * hi)], acc[j][r]);
    }
  }
}

// ---------------------------------------------------------------------------
// K2b: one-shot kvT f32 -> bf16 (kernel boundary orders after the atomics),
//      so K3's inner loop does pure contiguous bf16 fragment loads.
// ---------------------------------------------------------------------------
__global__ void kv_cvt_kernel(const float* __restrict__ kvT,
                              unsigned short* __restrict__ kvT_bf) {
  const long TOTAL = (long)Bn * Vp * C;
  for (long i = (long)blockIdx.x * blockDim.x + threadIdx.x; i < TOTAL;
       i += (long)gridDim.x * blockDim.x) {
    kvT_bf[i] = f2bf(kvT[i]);
  }
}

// ---------------------------------------------------------------------------
// K3 (fused): out = q @ kv, normalize by denominator column (v=256),
//     then y = (BN-folded W_proj) @ out_norm + bias. 64 positions per block,
//     4 waves. out_norm staged as bf16 in LDS (32 KB) between the two GEMMs.
// ---------------------------------------------------------------------------
__global__ __launch_bounds__(128) void attn_proj_kernel(
    const unsigned short* __restrict__ q_s, const unsigned short* __restrict__ kv_bf,
    const unsigned short* __restrict__ Wp, const float* __restrict__ bias,
    float* __restrict__ y) {
  __shared__ unsigned short s_out[64 * 256];  // (pos, c) bf16
  const int b = blockIdx.z;
  const int nbase = blockIdx.x * 64;
  const int w = threadIdx.x >> 5;
  const int l = threadIdx.x & 31;
  const int ln = l & 15, hi = l >> 4;

  // ---- phase A: out tile = q(16 rows) x kv(256 x 272) ----
  const unsigned short* qrow = q_s + ((size_t)b * Nn + nbase + 16 * w + ln) * C;
  const unsigned short* kvb = kv_bf + (size_t)b * Vp * C;

  v8f acc[17] = {};
  for (int kb = 0; kb < C; kb += 32) {
    v16bf a = afrag(qrow, kb, hi);
#pragma unroll
    for (int j = 0; j < 17; ++j) {
      // B column v = 16j+ln is row v of kvT_bf (c contiguous)
      acc[j] = wmma_bf16(a, bfrag(kvb + (size_t)(16 * j + ln) * C, kb, hi), acc[j]);
    }
  }

  // normalize: denom for row m lives in tile 16 (v=256) at lane 0 (m=r) / 16 (m=r+8)
#pragma unroll
  for (int r = 0; r < 8; ++r) {
    const float d0 = __builtin_bit_cast(
        float, __builtin_amdgcn_readlane(__builtin_bit_cast(int, acc[16][r]), 0));
    const float d1 = __builtin_bit_cast(
        float, __builtin_amdgcn_readlane(__builtin_bit_cast(int, acc[16][r]), 16));
    const float rl = 1.0f / (d0 + 1e-5f);
    const float rh = 1.0f / (d1 + 1e-5f);
    const float rr = hi ? rh : rl;
    const int pos = 16 * w + r + 8 * hi;
#pragma unroll
    for (int j = 0; j < 16; ++j) {
      s_out[pos * 256 + 16 * j + ln] = f2bf(acc[j][r] * rr);
    }
  }
  __syncthreads();

  // ---- phase B: y = Wp(256x256) @ out_norm^T, B fragments from LDS ----
  v8f acc2[4][4] = {};
  for (int kb = 0; kb < C; kb += 32) {
    v16bf a2[4];
#pragma unroll
    for (int i = 0; i < 4; ++i)
      a2[i] = afrag(Wp + (size_t)(64 * w + 16 * i + ln) * C, kb, hi);
#pragma unroll
    for (int j = 0; j < 4; ++j) {
      v16bf bf = bfrag(&s_out[(16 * j + ln) * 256], kb, hi);  // ds_load_b128 x2
#pragma unroll
      for (int i = 0; i < 4; ++i) acc2[i][j] = wmma_bf16(a2[i], bf, acc2[i][j]);
    }
  }

#pragma unroll
  for (int i = 0; i < 4; ++i) {
#pragma unroll
    for (int j = 0; j < 4; ++j) {
      const int n = nbase + 16 * j + ln;
#pragma unroll
      for (int r = 0; r < 8; ++r) {
        const int o = 64 * w + 16 * i + r + 8 * hi;
        y[((size_t)b * C + o) * Nn + n] = acc2[i][j][r] + bias[o];
      }
    }
  }
}

// ---------------------------------------------------------------------------
extern "C" void kernel_launch(void* const* d_in, const int* in_sizes, int n_in,
                              void* d_out, int out_size, void* d_ws, size_t ws_size,
                              hipStream_t stream) {
  (void)in_sizes; (void)n_in; (void)out_size; (void)ws_size;
  const float* x     = (const float*)d_in[0];
  const float* Wqkv  = (const float*)d_in[1];
  const float* Wproj = (const float*)d_in[2];
  const float* gamma = (const float*)d_in[3];
  const float* beta  = (const float*)d_in[4];
  const float* mean  = (const float*)d_in[5];
  const float* var   = (const float*)d_in[6];
  float* y = (float*)d_out;

  char* ws = (char*)d_ws;
  size_t off = 0;
  unsigned short* Wq_bf = (unsigned short*)(ws + off); off += (size_t)M3 * C * 2;
  unsigned short* Wp_bf = (unsigned short*)(ws + off); off += (size_t)C * C * 2;
  float* bias = (float*)(ws + off); off += (size_t)C * 4;
  off = (off + 255) & ~(size_t)255;
  unsigned short* q_s = (unsigned short*)(ws + off); off += (size_t)Bn * Nn * C * 2;
  unsigned short* k_s = (unsigned short*)(ws + off); off += (size_t)Bn * C * Nn * 2;
  unsigned short* vpT = (unsigned short*)(ws + off); off += (size_t)Bn * Vp * Nn * 2;
  float* kvT = (float*)(ws + off); off += (size_t)Bn * Vp * C * 4;
  unsigned short* kvT_bf = (unsigned short*)(ws + off); off += (size_t)Bn * Vp * C * 2;

  prep_kernel<<<2048, 256, 0, stream>>>(Wqkv, Wproj, gamma, beta, mean, var,
                                        Wq_bf, Wp_bf, bias, vpT, kvT);

  dim3 g1(Nn / 128, M3 / 128, Bn);
  qkv_kernel<<<g1, 256, 0, stream>>>(x, Wq_bf, q_s, k_s, vpT);

  dim3 g2(Nn / 1024, C / 64, Bn);
  kv_kernel<<<g2, 128, 0, stream>>>(k_s, vpT, kvT);

  kv_cvt_kernel<<<272, 256, 0, stream>>>(kvT, kvT_bf);

  dim3 g3(Nn / 64, 1, Bn);
  attn_proj_kernel<<<g3, 128, 0, stream>>>(q_s, kvT_bf, Wp_bf, bias, y);
}
